// SpatialSelfAttention_3324304687129
// MI455X (gfx1250) — matrix operand order repaired and checked by reading, not compiled, not run on hardware
//
#include <hip/hip_runtime.h>

typedef unsigned short u16;
typedef unsigned int   u32;
typedef unsigned long long u64;
typedef __attribute__((ext_vector_type(16))) __bf16 v16bf;
typedef __attribute__((ext_vector_type(8)))  float  v8f;

#define NB     2
#define NC     512
#define NHW    4096
#define NHEADS 8
#define DHEAD  64
#define NGRP   32
#define CPG    16
#define GELEM  (CPG*NHW)

// ---------- helpers ----------
static __device__ __forceinline__ u16 f2bf(float f) {
  union { float f; u32 u; } x; x.f = f;
  u32 r = x.u + 0x7fffu + ((x.u >> 16) & 1u);   // RNE
  return (u16)(r >> 16);
}

union FragU { v16bf v; uint4 q[2]; };
union Pack8 { u16 s[8]; uint4 q; };

// A-matrix 16x32 bf16 fragment: row M = lane&15; lanes<16 hold K {0..7,16..23},
// lanes>=16 hold K {8..15,24..31}. Source row-major [M][K].
static __device__ __forceinline__ v16bf load_a16x32(const u16* p, int ld) {
  int lane = threadIdx.x & 31;
  const u16* r = p + (size_t)(lane & 15) * ld + ((lane >> 4) << 3);
  FragU f;
  f.q[0] = *(const uint4*)(r);
  f.q[1] = *(const uint4*)(r + 16);
  return f.v;
}

// B-matrix 32x16 bf16 fragment: col N = lane&15; lanes<16 K {0..7,16..23}, lanes>=16
// K {8..15,24..31} (K pairs per VGPR). Source stored transposed (row=N, K contiguous).
static __device__ __forceinline__ v16bf load_b32x16_t(const u16* p, int ld) {
  int lane = threadIdx.x & 31;
  const u16* r = p + (size_t)(lane & 15) * ld + ((lane >> 4) << 3);
  FragU f;
  f.q[0] = *(const uint4*)(r);
  f.q[1] = *(const uint4*)(r + 16);
  return f.v;
}

#define WMMA_BF16(a, b, c) \
  __builtin_amdgcn_wmma_f32_16x16x32_bf16(false, (a), false, (b), (short)0, (c), false, false)

// ---------- Tensor Data Mover (CDNA5 TDM) ----------
#if __has_builtin(__builtin_amdgcn_tensor_load_to_lds)
#define HAVE_TDM 1
typedef __attribute__((ext_vector_type(4))) u32 v4u;
typedef __attribute__((ext_vector_type(8))) int v8i;
typedef __attribute__((ext_vector_type(4))) int v4i;

// 2D tile load: rows x we 16-bit elements, row stride `stride` elements.
static __device__ __forceinline__ void tdm_load_2d(u32 lds_addr, const u16* gp,
                                                   u32 we, u32 rows, u32 stride) {
  u64 ga = (u64)(size_t)gp;
  v4u g0;
  g0[0] = 1u;                                            // count=1, user mode
  g0[1] = lds_addr;                                      // LDS byte address
  g0[2] = (u32)(ga & 0xffffffffu);                       // global_addr[31:0]
  g0[3] = (u32)((ga >> 32) & 0x01ffffffu) | (2u << 30);  // addr[56:32] | type=2
  v8i g1;
  g1[0] = (int)(1u << 16);                               // data_size=1 (2B), mask=0
  g1[1] = (int)((we & 0xffffu) << 16);                   // tensor_dim0 lo16
  g1[2] = (int)(((we >> 16) & 0xffffu) | ((rows & 0xffffu) << 16)); // dim0 hi | dim1 lo
  g1[3] = (int)(((rows >> 16) & 0xffffu) | ((we & 0xffffu) << 16)); // dim1 hi | tile_dim0
  g1[4] = (int)(rows & 0xffffu);                         // tile_dim1 ; tile_dim2=0
  g1[5] = (int)stride;                                   // tensor_dim0_stride lo32
  g1[6] = 0;
  g1[7] = 0;
  v4i zz = {0, 0, 0, 0};
#if __clang_major__ >= 23
  v8i z8 = {0, 0, 0, 0, 0, 0, 0, 0};
  __builtin_amdgcn_tensor_load_to_lds(g0, g1, zz, zz, z8, 0);
#else
  __builtin_amdgcn_tensor_load_to_lds(g0, g1, zz, zz, 0);
#endif
}
#endif

// ---------- kernel 1: GroupNorm statistics ----------
__global__ __launch_bounds__(256) void k_gn_stats(const float* __restrict__ x,
                                                  float* __restrict__ stats) {
  int bg = blockIdx.x;
  const float4* p = (const float4*)(x + (size_t)bg * GELEM);
  float s1 = 0.f, s2 = 0.f;
  for (int i = threadIdx.x; i < GELEM / 4; i += 256) {
    float4 v = p[i];
    s1 += v.x + v.y + v.z + v.w;
    s2 += v.x * v.x + v.y * v.y + v.z * v.z + v.w * v.w;
  }
  __shared__ float r1[256], r2[256];
  r1[threadIdx.x] = s1; r2[threadIdx.x] = s2;
  __syncthreads();
  for (int off = 128; off > 0; off >>= 1) {
    if (threadIdx.x < off) {
      r1[threadIdx.x] += r1[threadIdx.x + off];
      r2[threadIdx.x] += r2[threadIdx.x + off];
    }
    __syncthreads();
  }
  if (threadIdx.x == 0) {
    float mean = r1[0] * (1.f / GELEM);
    float var  = r2[0] * (1.f / GELEM) - mean * mean;
    stats[2 * bg]     = mean;
    stats[2 * bg + 1] = rsqrtf(var + 1e-6f);
  }
}

// ---------- kernel 2: cast weights to bf16 ----------
__global__ __launch_bounds__(256) void k_cast_w(const float* __restrict__ qkv_w,
                                                const float* __restrict__ out_w,
                                                u16* __restrict__ wq,
                                                u16* __restrict__ wo) {
  const int NQ = 3 * NC * NC, NO = NC * NC;
  for (int i = blockIdx.x * 256 + threadIdx.x; i < NQ + NO; i += gridDim.x * 256) {
    if (i < NQ) wq[i] = f2bf(qkv_w[i]);
    else        wo[i - NQ] = f2bf(out_w[i - NQ]);
  }
}

// ---------- kernel 3: GN apply + transpose -> hn[b][n][c] bf16 ----------
__global__ __launch_bounds__(256) void k_gn_apply(const float* __restrict__ x,
                                                  const float* __restrict__ gw,
                                                  const float* __restrict__ gb,
                                                  const float* __restrict__ stats,
                                                  u16* __restrict__ hn) {
  __shared__ u16 t[32][33];
  int b  = blockIdx.z;
  int c0 = blockIdx.x * 32;
  int n0 = blockIdx.y * 32;
  #pragma unroll
  for (int i = 0; i < 4; i++) {
    int idx = threadIdx.x + 256 * i;
    int cl = idx >> 5, nl = idx & 31;
    int c = c0 + cl;
    float mean = stats[2 * (b * NGRP + (c >> 4))];
    float rstd = stats[2 * (b * NGRP + (c >> 4)) + 1];
    float v = x[((size_t)(b * NC + c)) * NHW + n0 + nl];
    t[cl][nl] = f2bf((v - mean) * rstd * gw[c] + gb[c]);
  }
  __syncthreads();
  #pragma unroll
  for (int i = 0; i < 4; i++) {
    int idx = threadIdx.x + 256 * i;
    int cl = idx & 31, nl = idx >> 5;
    hn[((size_t)(b * NHW + n0 + nl)) * NC + c0 + cl] = t[cl][nl];
  }
}

// ---------- kernel 4: QKV GEMM (LDS-staged B, double-buffered) ----------
__global__ __launch_bounds__(256) void k_qkv_gemm(const u16* __restrict__ W,
                                                  const u16* __restrict__ hn,
                                                  const float* __restrict__ bias,
                                                  u16* __restrict__ Q,
                                                  u16* __restrict__ K,
                                                  u16* __restrict__ V) {
  __shared__ u16 Bt[2][64][32];              // 8 KB, shared by all 8 waves
  int wave = threadIdx.x >> 5, lane = threadIdx.x & 31;
  int m0 = blockIdx.x * 128 + wave * 16;
  int n0 = blockIdx.y * 64;
  int b  = blockIdx.z;
  const u16* HN = hn + (size_t)b * NHW * NC;

#ifdef HAVE_TDM
  auto stage_issue = [&](int kk, int bsel) {
    if (wave == 0)
      tdm_load_2d((u32)(size_t)&Bt[bsel][0][0], HN + (size_t)n0 * NC + kk, 32, 64, NC);
  };
  auto stage_commit = [&](int bsel) {
    (void)bsel;
    if (wave == 0) __builtin_amdgcn_s_wait_tensorcnt(0);
  };
#else
  uint4 breg;
  auto stage_issue = [&](int kk, int bsel) {
    (void)bsel;
    int row = threadIdx.x >> 2, q = threadIdx.x & 3;
    breg = *(const uint4*)(HN + (size_t)(n0 + row) * NC + kk + q * 8);
  };
  auto stage_commit = [&](int bsel) {
    int row = threadIdx.x >> 2, q = threadIdx.x & 3;
    *(uint4*)(&Bt[bsel][row][q * 8]) = breg;
  };
#endif

  v8f z = {0.f,0.f,0.f,0.f,0.f,0.f,0.f,0.f};
  v8f acc[4] = {z, z, z, z};

  stage_issue(0, 0);
  stage_commit(0);
  __syncthreads();
  v16bf a_cur = load_a16x32(W + (size_t)m0 * NC, NC);

  for (int k = 0; k < NC; k += 32) {
    int cur = (k >> 5) & 1, nxt = cur ^ 1;
    bool more = (k + 32) < NC;
    v16bf a_nxt;
    if (more) {
      a_nxt = load_a16x32(W + (size_t)m0 * NC + k + 32, NC);
      stage_issue(k + 32, nxt);
    }
    #pragma unroll
    for (int g = 0; g < 4; g++) {
      v16bf bf = load_b32x16_t(&Bt[cur][16 * g][0], 32);
      acc[g] = WMMA_BF16(a_cur, bf, acc[g]);
    }
    if (more) { stage_commit(nxt); a_cur = a_nxt; }
    __syncthreads();
  }

  int hh = lane >> 4, nl = lane & 15;
  float bi[8];
  #pragma unroll
  for (int r = 0; r < 8; r++) bi[r] = bias[m0 + r + 8 * hh];

  if (m0 < NC) {                                           // ---- Q ----
    int h = m0 >> 6, d0 = (m0 & 63) + 8 * hh;
    u16* dst = Q + ((size_t)(b * NHEADS + h)) * NHW * DHEAD;
    #pragma unroll
    for (int g = 0; g < 4; g++) {
      int n = n0 + 16 * g + nl;
      Pack8 pk;
      #pragma unroll
      for (int r = 0; r < 8; r++) pk.s[r] = f2bf(acc[g][r] + bi[r]);
      *(uint4*)(dst + (size_t)n * DHEAD + d0) = pk.q;
    }
  } else if (m0 < 2 * NC) {                                // ---- K ----
    int mo = m0 - NC;
    int h = mo >> 6, d0 = (mo & 63) + 8 * hh;
    u16* dst = K + ((size_t)(b * NHEADS + h)) * NHW * DHEAD;
    #pragma unroll
    for (int g = 0; g < 4; g++) {
      int n = n0 + 16 * g + nl;
      Pack8 pk;
      #pragma unroll
      for (int r = 0; r < 8; r++) pk.s[r] = f2bf(acc[g][r] + bi[r]);
      *(uint4*)(dst + (size_t)n * DHEAD + d0) = pk.q;
    }
  } else {                                                 // ---- V ----
    int mo = m0 - 2 * NC;
    int h = mo >> 6, d0 = (mo & 63) + 8 * hh;
    u16* dst = V + ((size_t)(b * NHEADS + h)) * DHEAD * NHW;
    #pragma unroll
    for (int g = 0; g < 4; g++) {
      int n = n0 + 16 * g + nl;
      #pragma unroll
      for (int r = 0; r < 8; r++)
        dst[(size_t)(d0 + r) * NHW + n] = f2bf(acc[g][r] + bi[r]);
    }
  }
}

// ---------- kernel 5: flash attention, block-shared K/V tiles in LDS ----------
__global__ __launch_bounds__(256) void k_flash(const u16* __restrict__ Q,
                                               const u16* __restrict__ K,
                                               const u16* __restrict__ V,
                                               u16* __restrict__ O) {
  __shared__ u16 Kt[2][64][64];              // 16 KB  (keys x d)
  __shared__ u16 Vt[2][64][64];              // 16 KB  (d x keys)
  __shared__ u16 plds[8][16][64];            // 16 KB  per-wave P transpose
  int wave = threadIdx.x >> 5, lane = threadIdx.x & 31;
  int hh = lane >> 4, nl = lane & 15;
  int bh = blockIdx.y;
  int q0 = (blockIdx.x * 8 + wave) * 16;
  const u16* Qh = Q + (size_t)bh * NHW * DHEAD;
  const u16* Kh = K + (size_t)bh * NHW * DHEAD;
  const u16* Vh = V + (size_t)bh * DHEAD * NHW;

#ifdef HAVE_TDM
  auto stage_issue = [&](int ktN, int bsel) {
    if (wave == 0) {
      tdm_load_2d((u32)(size_t)&Kt[bsel][0][0], Kh + (size_t)ktN * DHEAD, DHEAD, 64, DHEAD);
      tdm_load_2d((u32)(size_t)&Vt[bsel][0][0], Vh + ktN, 64, DHEAD, NHW);
    }
  };
  auto stage_commit = [&](int bsel) {
    (void)bsel;
    if (wave == 0) __builtin_amdgcn_s_wait_tensorcnt(0);
  };
#else
  uint4 kreg[2], vreg[2];
  auto stage_issue = [&](int ktN, int bsel) {
    (void)bsel;
    #pragma unroll
    for (int j = 0; j < 2; j++) {
      int idx = threadIdx.x + 256 * j;
      int row = idx >> 3, q = idx & 7;
      kreg[j] = *(const uint4*)(Kh + (size_t)(ktN + row) * DHEAD + q * 8);
      vreg[j] = *(const uint4*)(Vh + (size_t)row * NHW + ktN + q * 8);
    }
  };
  auto stage_commit = [&](int bsel) {
    #pragma unroll
    for (int j = 0; j < 2; j++) {
      int idx = threadIdx.x + 256 * j;
      int row = idx >> 3, q = idx & 7;
      *(uint4*)(&Kt[bsel][row][q * 8]) = kreg[j];
      *(uint4*)(&Vt[bsel][row][q * 8]) = vreg[j];
    }
  };
#endif

  v16bf a0 = load_a16x32(Qh + (size_t)q0 * DHEAD,      DHEAD);
  v16bf a1 = load_a16x32(Qh + (size_t)q0 * DHEAD + 32, DHEAD);

  v8f z = {0.f,0.f,0.f,0.f,0.f,0.f,0.f,0.f};
  v8f acc[4] = {z, z, z, z};
  float mrow[8], lrow[8];
  #pragma unroll
  for (int r = 0; r < 8; r++) { mrow[r] = -3.0e38f; lrow[r] = 0.f; }
  const float SCL = 0.125f * 1.44269504089f;  // 1/sqrt(64) * log2(e)

  stage_issue(0, 0);
  stage_commit(0);
  __syncthreads();

  for (int kt = 0; kt < NHW; kt += 64) {
    int cur = (kt >> 6) & 1, nxt = cur ^ 1;
    bool more = (kt + 64) < NHW;
    if (more) stage_issue(kt + 64, nxt);

    // ---- S = Q^T K  (8 WMMAs, B-frags from LDS) ----
    v8f s[4];
    #pragma unroll
    for (int g = 0; g < 4; g++) {
      s[g] = z;
      v16bf b0 = load_b32x16_t(&Kt[cur][16 * g][0],      64);
      s[g] = WMMA_BF16(a0, b0, s[g]);
      v16bf b1 = load_b32x16_t(&Kt[cur][16 * g][0] + 32, 64);
      s[g] = WMMA_BF16(a1, b1, s[g]);
    }
    // ---- online softmax ----
    #pragma unroll
    for (int r = 0; r < 8; r++) {
      float mx = fmaxf(fmaxf(s[0][r], s[1][r]), fmaxf(s[2][r], s[3][r]));
      mx = fmaxf(mx, __shfl_xor(mx, 1));
      mx = fmaxf(mx, __shfl_xor(mx, 2));
      mx = fmaxf(mx, __shfl_xor(mx, 4));
      mx = fmaxf(mx, __shfl_xor(mx, 8));
      float mn = fmaxf(mrow[r], mx);
      float alpha = exp2f((mrow[r] - mn) * SCL);
      mrow[r] = mn;
      lrow[r] *= alpha;
      acc[0][r] *= alpha; acc[1][r] *= alpha;
      acc[2][r] *= alpha; acc[3][r] *= alpha;
      float ps = 0.f;
      #pragma unroll
      for (int g = 0; g < 4; g++) {
        float p = exp2f((s[g][r] - mn) * SCL);
        s[g][r] = p;
        ps += p;
      }
      ps += __shfl_xor(ps, 1);
      ps += __shfl_xor(ps, 2);
      ps += __shfl_xor(ps, 4);
      ps += __shfl_xor(ps, 8);
      lrow[r] += ps;
    }
    // ---- P: D-layout -> A-layout via wave-private LDS strip ----
    u16 (*pl)[64] = plds[wave];
    #pragma unroll
    for (int g = 0; g < 4; g++)
      #pragma unroll
      for (int r = 0; r < 8; r++)
        pl[r + 8 * hh][16 * g + nl] = f2bf(s[g][r]);
    v16bf p0 = load_a16x32(&pl[0][0],      64);
    v16bf p1 = load_a16x32(&pl[0][0] + 32, 64);
    // ---- O += P * V^T (8 WMMAs, B-frags from LDS) ----
    #pragma unroll
    for (int g = 0; g < 4; g++) {
      v16bf b0 = load_b32x16_t(&Vt[cur][16 * g][0],      64);
      acc[g] = WMMA_BF16(p0, b0, acc[g]);
      v16bf b1 = load_b32x16_t(&Vt[cur][16 * g][0] + 32, 64);
      acc[g] = WMMA_BF16(p1, b1, acc[g]);
    }
    if (more) stage_commit(nxt);
    __syncthreads();
  }

  // ---- normalize + store O[b][n][h*64+d] ----
  int b = bh >> 3, h = bh & 7;
  #pragma unroll
  for (int r = 0; r < 8; r++) {
    float rl = 1.f / lrow[r];
    int n = q0 + r + 8 * hh;
    #pragma unroll
    for (int g = 0; g < 4; g++) {
      int c = h * DHEAD + 16 * g + nl;
      O[((size_t)(b * NHW + n)) * NC + c] = f2bf(acc[g][r] * rl);
    }
  }
}

// ---------- kernel 6: out projection + bias + residual ----------
__global__ __launch_bounds__(256) void k_out_proj(const u16* __restrict__ W,
                                                  const u16* __restrict__ O,
                                                  const float* __restrict__ bias,
                                                  const float* __restrict__ x,
                                                  float* __restrict__ out) {
  __shared__ u16 Bt[2][64][32];
  int wave = threadIdx.x >> 5, lane = threadIdx.x & 31;
  int m0 = blockIdx.x * 128 + wave * 16;
  int n0 = blockIdx.y * 64;
  int b  = blockIdx.z;
  const u16* Ob = O + (size_t)b * NHW * NC;

#ifdef HAVE_TDM
  auto stage_issue = [&](int kk, int bsel) {
    if (wave == 0)
      tdm_load_2d((u32)(size_t)&Bt[bsel][0][0], Ob + (size_t)n0 * NC + kk, 32, 64, NC);
  };
  auto stage_commit = [&](int bsel) {
    (void)bsel;
    if (wave == 0) __builtin_amdgcn_s_wait_tensorcnt(0);
  };
#else
  uint4 breg;
  auto stage_issue = [&](int kk, int bsel) {
    (void)bsel;
    int row = threadIdx.x >> 2, q = threadIdx.x & 3;
    breg = *(const uint4*)(Ob + (size_t)(n0 + row) * NC + kk + q * 8);
  };
  auto stage_commit = [&](int bsel) {
    int row = threadIdx.x >> 2, q = threadIdx.x & 3;
    *(uint4*)(&Bt[bsel][row][q * 8]) = breg;
  };
#endif

  v8f z = {0.f,0.f,0.f,0.f,0.f,0.f,0.f,0.f};
  v8f acc[4] = {z, z, z, z};

  stage_issue(0, 0);
  stage_commit(0);
  __syncthreads();
  v16bf a_cur = load_a16x32(W + (size_t)m0 * NC, NC);

  for (int k = 0; k < NC; k += 32) {
    int cur = (k >> 5) & 1, nxt = cur ^ 1;
    bool more = (k + 32) < NC;
    v16bf a_nxt;
    if (more) {
      a_nxt = load_a16x32(W + (size_t)m0 * NC + k + 32, NC);
      stage_issue(k + 32, nxt);
    }
    #pragma unroll
    for (int g = 0; g < 4; g++) {
      v16bf bf = load_b32x16_t(&Bt[cur][16 * g][0], 32);
      acc[g] = WMMA_BF16(a_cur, bf, acc[g]);
    }
    if (more) { stage_commit(nxt); a_cur = a_nxt; }
    __syncthreads();
  }

  int hh = lane >> 4, nl = lane & 15;
  #pragma unroll
  for (int r = 0; r < 8; r++) {
    int co = m0 + r + 8 * hh;
    float bi = bias[co];
    #pragma unroll
    for (int g = 0; g < 4; g++) {
      int n = n0 + 16 * g + nl;
      size_t idx = ((size_t)(b * NC + co)) * NHW + n;
      out[idx] = x[idx] + acc[g][r] + bi;
    }
  }
}

// ---------- launcher ----------
extern "C" void kernel_launch(void* const* d_in, const int* in_sizes, int n_in,
                              void* d_out, int out_size, void* d_ws, size_t ws_size,
                              hipStream_t stream) {
  (void)in_sizes; (void)n_in; (void)out_size; (void)ws_size;
  const float* x     = (const float*)d_in[0];
  const float* gn_w  = (const float*)d_in[1];
  const float* gn_b  = (const float*)d_in[2];
  const float* qkv_w = (const float*)d_in[3];
  const float* qkv_b = (const float*)d_in[4];
  const float* out_w = (const float*)d_in[5];
  const float* out_b = (const float*)d_in[6];
  float* out = (float*)d_out;

  char* w = (char*)d_ws;
  u16*  hn    = (u16*)(w + 0);                    //  8 MB  [b][n][c]
  u16*  wq    = (u16*)(w + (8u  << 20));          //  1.5MB
  u16*  wo    = (u16*)(w + (10u << 20));          //  0.5MB
  u16*  Qb    = (u16*)(w + (11u << 20));          //  8 MB  [b][h][n][d]
  u16*  Kb    = (u16*)(w + (19u << 20));          //  8 MB  [b][h][n][d]
  u16*  Vb    = (u16*)(w + (27u << 20));          //  8 MB  [b][h][d][n]
  u16*  Ob    = (u16*)(w + (35u << 20));          //  8 MB  [b][n][c]
  float* stats = (float*)(w + (43u << 20));       //  512 B

  k_gn_stats<<<NB * NGRP, 256, 0, stream>>>(x, stats);
  k_cast_w<<<1024, 256, 0, stream>>>(qkv_w, out_w, wq, wo);
  k_gn_apply<<<dim3(NC / 32, NHW / 32, NB), 256, 0, stream>>>(x, gn_w, gn_b, stats, hn);
  k_qkv_gemm<<<dim3(3 * NC / 128, NHW / 64, NB), 256, 0, stream>>>(wq, hn, qkv_b, Qb, Kb, Vb);
  k_flash<<<dim3(NHW / (8 * 16), NB * NHEADS), 256, 0, stream>>>(Qb, Kb, Vb, Ob);
  k_out_proj<<<dim3(NC / 128, NHW / 64, NB), 256, 0, stream>>>(wo, Ob, out_b, x, out);
}
